// TitleEncoder_56642028700226
// MI455X (gfx1250) — matrix operand order repaired and checked by reading, not compiled, not run on hardware
//
#include <hip/hip_runtime.h>
#include <hip/hip_bf16.h>
#include <cstdint>

// ---- CDNA5 (gfx1250) wave32 WMMA types ----
typedef __bf16 bf16;
typedef bf16  v16bf __attribute__((ext_vector_type(16)));
typedef bf16  v8bf  __attribute__((ext_vector_type(8)));
typedef float v8f   __attribute__((ext_vector_type(8)));

union AFrag { v16bf v; bf16 e[16]; v8bf h[2]; };

// Problem constants
#define VOCAB 50000
#define EMB   300
#define EPAD  320      // E padded so K = 3*EPAD = 960 = 30*32
#define FILT  400
#define ATT   200
#define APAD  208      // ATT padded to 13*16
#define KW    3
#define BB    4096
#define NN    30
#define MM    (BB*NN)          // 122880
#define KCONV 960              // 30 chunks of 32
#define KATT  416              // 13 chunks of 32 (400 padded)

// ---------------- prep kernels: repack weights to bf16 B-operand layout ----
// W2[f][k], k = kw*EPAD + e  (row-major, 960 per row) so each lane's B-frag
// slice (16 contiguous K for its column) is one 32-byte contiguous read.
__global__ void k_prep_w2(const float* __restrict__ conv_w, bf16* __restrict__ W2) {
    int idx = blockIdx.x * 256 + threadIdx.x;            // 400*960 = 384000
    if (idx >= FILT * KCONV) return;
    int f = idx / KCONV, k = idx % KCONV;
    int kw = k / EPAD, e = k % EPAD;
    float val = (e < EMB) ? conv_w[(size_t)f * (EMB * KW) + (size_t)e * KW + kw] : 0.0f;
    W2[idx] = (bf16)val;
}

// V2[a][k]: a = attention col (padded to 208), k over FILT padded to 416.
__global__ void k_prep_v2(const float* __restrict__ v, bf16* __restrict__ V2) {
    int idx = blockIdx.x * 256 + threadIdx.x;            // 208*416 = 86528
    if (idx >= APAD * KATT) return;
    int a = idx / KATT, k = idx % KATT;
    float val = (a < ATT && k < FILT) ? v[(size_t)k * ATT + a] : 0.0f;
    V2[idx] = (bf16)val;
}

__global__ void k_prep_q(const float* __restrict__ q, float* __restrict__ qpad) {
    int i = threadIdx.x;
    if (i < APAD) qpad[i] = (i < ATT) ? q[i] : 0.0f;
}

// ---------------- K1: fused gather + im2col conv GEMM + ReLU + attention ----
// block = 5 waves (160 thr) owns 16 rows of M; wave w does F-tiles 5w..5w+4.
// A strip (16 x 960 im2col, bf16) is staged in LDS once per block so the
// 150-WMMA hot loop is pure ds_load + global_load + v_wmma (no branches).
__global__ __launch_bounds__(160)
void k_conv_att(const int*   __restrict__ enc,
                const float* __restrict__ emb,
                const float* __restrict__ conv_b,
                const float* __restrict__ vb,
                const bf16*  __restrict__ W2,
                const bf16*  __restrict__ V2,
                const float* __restrict__ qpad,
                bf16*  __restrict__ Cg,
                float* __restrict__ score) {
    __shared__ __align__(16) bf16 As[16 * KCONV];   // 30,720 B
    __shared__ __align__(16) bf16 Cs[16 * KATT];    // 13,312 B
    __shared__ float sscore[16];

    const int tid  = threadIdx.x;
    const int lane = tid & 31;
    const int wave = tid >> 5;
    const int r    = lane & 15;       // row (M) within tile / column within tile
    const int hs   = lane >> 4;       // half-select (A/B lane group)
    const int m0   = blockIdx.x * 16;

    // ---- stage A strip: As[row][kw*320+e] = emb[enc[b, n+kw-1], e] (masked)
    for (int idx = tid; idx < 16 * KCONV; idx += 160) {
        int row = idx / KCONV;
        int k   = idx % KCONV;
        int p   = k / EPAD;            // conv tap 0..2
        int e   = k % EPAD;
        int m   = m0 + row;
        int b   = m / NN;
        int n   = m % NN;
        int n_src = n + p - 1;
        float val = 0.0f;
        if (n_src >= 0 && n_src < NN && e < EMB) {
            int token = enc[b * NN + n_src];
            val = emb[(size_t)token * EMB + e];
        }
        As[idx] = (bf16)val;
    }
    if (tid < 16) sscore[tid] = 0.0f;
    for (int idx = tid; idx < 16 * 16; idx += 160)        // zero pad Cs cols 400..415
        Cs[(idx >> 4) * KATT + FILT + (idx & 15)] = (bf16)0.0f;
    __syncthreads();

    // ---- phase 1: C[16 x 400] = relu(As @ W2^T + conv_b) ----
    v8f acc[5];
#pragma unroll
    for (int t = 0; t < 5; ++t) {
        int f = (wave * 5 + t) * 16 + r;
        float bias = conv_b[f];
#pragma unroll
        for (int i = 0; i < 8; ++i) acc[t][i] = bias;
    }

    for (int j = 0; j < 30; ++j) {                 // K chunks of 32
        AFrag af;
        // A layout (matches B): lane<16 -> K {0..15}; lane>=16 -> {16..31}
        const bf16* ap = &As[r * KCONV + j * 32];
        af.h[0] = *reinterpret_cast<const v8bf*>(ap +     (hs ? 16 : 0));
        af.h[1] = *reinterpret_cast<const v8bf*>(ap + 8 + (hs ? 16 : 0));
#pragma unroll
        for (int t = 0; t < 5; ++t) {
            int fcol = (wave * 5 + t) * 16 + r;
            const bf16* bp = W2 + (size_t)fcol * KCONV + j * 32 + (hs ? 16 : 0);
            AFrag bf_;
            bf_.h[0] = *reinterpret_cast<const v8bf*>(bp);
            bf_.h[1] = *reinterpret_cast<const v8bf*>(bp + 8);
            acc[t] = __builtin_amdgcn_wmma_f32_16x16x32_bf16(
                false, af.v, false, bf_.v, (short)0, acc[t], false, false);
        }
    }

    // ReLU; stage strip to LDS (bf16, K padded to 416) and to global C.
#pragma unroll
    for (int t = 0; t < 5; ++t) {
        int fcol = (wave * 5 + t) * 16 + r;
#pragma unroll
        for (int i = 0; i < 8; ++i) {
            int  row = i + (hs ? 8 : 0);                  // C/D layout
            float x  = acc[t][i];
            x = x > 0.0f ? x : 0.0f;
            bf16 hv = (bf16)x;
            Cs[row * KATT + fcol] = hv;
            Cg[(size_t)(m0 + row) * FILT + fcol] = hv;
        }
    }
    __syncthreads();

    // ---- phase 2: a = tanh(Cs @ V2^T + vb); score = a @ q (row reduce) ----
    for (int t = wave; t < 13; t += 5) {
        int colg = t * 16 + r;
        float bias = (colg < ATT) ? vb[colg] : 0.0f;
        v8f a2;
#pragma unroll
        for (int i = 0; i < 8; ++i) a2[i] = bias;
        for (int j = 0; j < 13; ++j) {
            AFrag af, bf_;
            const bf16* ap = &Cs[r * KATT + j * 32];
            af.h[0] = *reinterpret_cast<const v8bf*>(ap +     (hs ? 16 : 0));
            af.h[1] = *reinterpret_cast<const v8bf*>(ap + 8 + (hs ? 16 : 0));
            const bf16* bp = V2 + (size_t)colg * KATT + j * 32 + (hs ? 16 : 0);
            bf_.h[0] = *reinterpret_cast<const v8bf*>(bp);
            bf_.h[1] = *reinterpret_cast<const v8bf*>(bp + 8);
            a2 = __builtin_amdgcn_wmma_f32_16x16x32_bf16(
                false, af.v, false, bf_.v, (short)0, a2, false, false);
        }
        float qc = qpad[colg];
#pragma unroll
        for (int i = 0; i < 8; ++i) {
            int row = i + (hs ? 8 : 0);
            float p = tanhf(a2[i]) * qc;
            atomicAdd(&sscore[row], p);                   // ds_add_f32
        }
    }
    __syncthreads();
    if (tid < 16) score[m0 + tid] = sscore[tid];
}

// ---------------- K2: softmax over the BATCH dim (per position n) ----------
__global__ __launch_bounds__(256)
void k_softmax(const float* __restrict__ score, const int* __restrict__ enc,
               float* __restrict__ alpha) {
    const int n = blockIdx.x;          // 0..29
    const int tid = threadIdx.x;       // 256
    __shared__ float red[256];
    float mx = -3.4e38f;
    for (int b = tid; b < BB; b += 256) mx = fmaxf(mx, score[b * NN + n]);
    red[tid] = mx; __syncthreads();
    for (int s = 128; s > 0; s >>= 1) { if (tid < s) red[tid] = fmaxf(red[tid], red[tid + s]); __syncthreads(); }
    mx = red[0]; __syncthreads();
    float sum = 0.0f;
    for (int b = tid; b < BB; b += 256) sum += __expf(score[b * NN + n] - mx);
    red[tid] = sum; __syncthreads();
    for (int s = 128; s > 0; s >>= 1) { if (tid < s) red[tid] += red[tid + s]; __syncthreads(); }
    float inv = 1.0f / red[0];
    for (int b = tid; b < BB; b += 256) {
        float e = __expf(score[b * NN + n] - mx) * inv;
        if (enc[b * NN + n] == 0) e = 0.0f;               // mask after softmax
        alpha[b * NN + n] = e;
    }
}

// ---------------- K3: e[b,f] = sum_n alpha[b,n] * C[b,n,f] (C in L2) -------
__global__ __launch_bounds__(256)
void k_pool(const float* __restrict__ alpha, const bf16* __restrict__ Cg,
            float* __restrict__ out) {
    int b = blockIdx.x;
    int f = blockIdx.y * 256 + threadIdx.x;
    if (f >= FILT) return;
    float s = 0.0f;
#pragma unroll 5
    for (int n = 0; n < NN; ++n) {
        float a = alpha[b * NN + n];                      // uniform -> scalar load
        s += a * (float)Cg[(size_t)(b * NN + n) * FILT + f];
    }
    out[(size_t)b * FILT + f] = s;
}

// ---------------- launch ----------------------------------------------------
extern "C" void kernel_launch(void* const* d_in, const int* in_sizes, int n_in,
                              void* d_out, int out_size, void* d_ws, size_t ws_size,
                              hipStream_t stream) {
    const int*   enc    = (const int*)  d_in[0];
    const float* emb    = (const float*)d_in[1];
    const float* conv_w = (const float*)d_in[2];
    const float* conv_b = (const float*)d_in[3];
    const float* v      = (const float*)d_in[4];
    const float* vb     = (const float*)d_in[5];
    const float* q      = (const float*)d_in[6];
    float* out = (float*)d_out;

    char* ws = (char*)d_ws;
    bf16*  W2    = (bf16*)(ws + 0);                        //   768,000 B
    bf16*  V2    = (bf16*)(ws + 768000);                   //   173,056 B
    float* qpad  = (float*)(ws + 941056);                  //     1,024 B
    float* score = (float*)(ws + 942080);                  //   491,520 B
    float* alpha = (float*)(ws + 1433600);                 //   491,520 B
    bf16*  Cg    = (bf16*)(ws + 1925120);                  // 98,304,000 B (fits L2)

    k_prep_w2<<<(FILT * KCONV + 255) / 256, 256, 0, stream>>>(conv_w, W2);
    k_prep_v2<<<(APAD * KATT + 255) / 256, 256, 0, stream>>>(v, V2);
    k_prep_q <<<1, 256, 0, stream>>>(q, qpad);

    k_conv_att<<<MM / 16, 160, 0, stream>>>(enc, emb, conv_b, vb, W2, V2, qpad, Cg, score);
    k_softmax<<<NN, 256, 0, stream>>>(score, enc, alpha);
    k_pool<<<dim3(BB, (FILT + 255) / 256), 256, 0, stream>>>(alpha, Cg, out);
}
